// STGCN_Like_Baseline_4896262717770
// MI455X (gfx1250) — matrix operand order repaired and checked by reading, not compiled, not run on hardware
//
#include <hip/hip_runtime.h>
#include <hip/hip_bf16.h>
#include <hip/hip_fp16.h>
#include <stdint.h>

#define NN  50000
#define NE  800000
#define TS  12
#define HID 64
#define NT  (NN / 16)          // 3125 node tiles of 16

typedef _Float16 v16h __attribute__((ext_vector_type(16)));
typedef _Float16 v8h  __attribute__((ext_vector_type(8)));
typedef float    v8f  __attribute__((ext_vector_type(8)));

union ABfrag {
  v16h v;
  v8h  h2[2];
  uint4 u4[2];
};

// CDNA5 has native v_tanh_f32 (TRANS pipe, co-executes with XDL WMMA).
static __device__ __forceinline__ float fast_tanh(float x) {
#if __has_builtin(__builtin_amdgcn_tanhf)
  return __builtin_amdgcn_tanhf(x);
#else
  float e = __expf(2.0f * x);
  return (e - 1.0f) * __builtin_amdgcn_rcpf(e + 1.0f);
#endif
}
// sigmoid(x) = 0.5*tanh(x/2) + 0.5  -> one TRANS + one FMA (no IEEE divide)
static __device__ __forceinline__ float fast_sigmoid(float x) {
  return __builtin_fmaf(0.5f, fast_tanh(0.5f * x), 0.5f);
}

// ---------------------------------------------------------------------------
// GCN preprocessing kernels (work on linear pre-activation aggregate: 24 f32
// per node = 12 timesteps x 2 input channels; GCN matmul applied afterwards).
// ---------------------------------------------------------------------------

__global__ void k_init(float* __restrict__ deg, float* __restrict__ agg) {
  int i = blockIdx.x * blockDim.x + threadIdx.x;
  if (i < NN) deg[i] = 1.0f;                  // +1 self loop in reference
  if (i < NN * 24) agg[i] = 0.0f;
}

__global__ void k_degree(const long long* __restrict__ ei, float* __restrict__ deg) {
  int e = blockIdx.x * blockDim.x + threadIdx.x;
  if (e >= NE) return;
  atomicAdd(&deg[(int)ei[NE + e]], 1.0f);
}

__global__ void k_dinv(float* __restrict__ deg) {  // in place: deg -> rsqrt(deg)
  int i = blockIdx.x * blockDim.x + threadIdx.x;
  if (i < NN) deg[i] = rsqrtf(deg[i]);
}

// agg[dst] += (dinv[src]*dinv[dst]) * x[src]  for all 24 channels at once
__global__ void k_scatter(const long long* __restrict__ ei,
                          const float* __restrict__ dinv,
                          const float* __restrict__ x,
                          float* __restrict__ agg) {
  int idx = blockIdx.x * blockDim.x + threadIdx.x;
  if (idx >= NE * 24) return;
  int e = idx / 24;
  int c = idx - e * 24;
  int s = (int)ei[e];
  int d = (int)ei[NE + e];
  float w = dinv[s] * dinv[d];
  atomicAdd(&agg[d * 24 + c], w * x[s * 24 + c]);
}

// spatial[t][n][j] = relu( (agg + dinv^2 * x) @ W_gcn + b_gcn )   as f16
__global__ void k_spatial(const float* __restrict__ x,
                          const float* __restrict__ agg,
                          const float* __restrict__ dinv,
                          const float* __restrict__ Wg,   // (2,64) row-major
                          const float* __restrict__ bg,
                          _Float16* __restrict__ spatial) {
  int idx = blockIdx.x * blockDim.x + threadIdx.x;
  if (idx >= NN * TS * HID) return;
  int j = idx & 63;
  int r = idx >> 6;
  int t = r % TS;
  int n = r / TS;
  float di = dinv[n];
  float sn = di * di;
  int base = n * 24 + t * 2;
  float a0 = agg[base + 0] + sn * x[base + 0];
  float a1 = agg[base + 1] + sn * x[base + 1];
  float v = a0 * Wg[j] + a1 * Wg[64 + j] + bg[j];
  v = v > 0.0f ? v : 0.0f;
  spatial[((size_t)t * NN + n) * HID + j] = (_Float16)v;
}

// ---------------------------------------------------------------------------
// Pack W_ih / W_hh (192x64 f32, gi = x @ W^T) into the CDNA5 WMMA 16-bit
// B-matrix (32x16) lane layout as f16.  Index: (((mat*12+jt)*2+kc)*32+lane)*8+v
//   lane L: column j = 16*jt + (L%16)
//           K rows   = 32*kc + (L>=16 ? 8 : 0) + 2v + (v>=4 ? 8 : 0),+1   (B[k][j] = W[j][k])
// ---------------------------------------------------------------------------
__global__ void k_pack(const float* __restrict__ Wih,
                       const float* __restrict__ Whh,
                       uint32_t* __restrict__ packed) {
  int tid = blockIdx.x * blockDim.x + threadIdx.x;
  if (tid >= 12288) return;
  int v  = tid & 7;
  int t2 = tid >> 3;
  int lane = t2 & 31;  t2 >>= 5;
  int kc   = t2 & 1;   t2 >>= 1;
  int jt   = t2 % 12;
  int mat  = t2 / 12;
  const float* W = mat ? Whh : Wih;
  int j  = 16 * jt + (lane & 15);
  int kb = 32 * kc + ((lane >> 4) ? 8 : 0) + 2 * v + (v >= 4 ? 8 : 0);
  union { _Float16 h[2]; uint32_t u; } p;
  p.h[0] = (_Float16)W[j * 64 + kb];
  p.h[1] = (_Float16)W[j * 64 + kb + 1];
  packed[tid] = p.u;
}

static __device__ __forceinline__ v16h load_btile(const uint32_t* ldsW,
                                                  int mat, int jt, int kc, int lane) {
  ABfrag b;
  const uint32_t* p = ldsW + (((mat * 12 + jt) * 2 + kc) * 32 + lane) * 8;
  b.u4[0] = *(const uint4*)p;
  b.u4[1] = *(const uint4*)(p + 4);
  return b.v;
}

#define WMMA(a, b, c) \
  __builtin_amdgcn_wmma_f32_16x16x32_f16(false, (a), false, (b), (short)0, (c), false, false)

// ---------------------------------------------------------------------------
// Fused GRU over 12 steps + FC head.  One wave = 16 nodes; h lives in four
// f32 C/D fragments for the whole recurrence. Weights staged into LDS (48KB)
// with CDNA5 async LDS-DMA (ASYNCcnt), reused by every wave for all 12 steps.
// B tiles are double-buffered in two disjoint 4-tile register groups so the
// LDS round-trip for the next WMMA group is covered by the current group.
// ---------------------------------------------------------------------------
__global__ __launch_bounds__(256, 1) void k_gru(
    const _Float16* __restrict__ spatial,
    const uint32_t* __restrict__ packedW,
    const float* __restrict__ b_ih, const float* __restrict__ b_hh,
    const float* __restrict__ W_fc, const float* __restrict__ b_fc,
    float* __restrict__ out) {
  __shared__ __align__(16) uint32_t ldsW[12288];          // 48 KB packed B tiles
  __shared__ __align__(32) _Float16 hstage[8][16 * HID];  // 16 KB, per-wave 16x64

  const int tid = threadIdx.x;

  // Async global->LDS weight stage: 12 x b128 per thread, no VGPR round-trip.
#pragma unroll 1
  for (int base = tid * 4; base < 12288; base += 256 * 4) {
    uint32_t laddr = (uint32_t)(uintptr_t)(&ldsW[base]);
    uint64_t gaddr = (uint64_t)(uintptr_t)(packedW + base);
    asm volatile("global_load_async_to_lds_b128 %0, %1, off"
                 :: "v"(laddr), "v"(gaddr) : "memory");
  }
  asm volatile("s_wait_asynccnt 0" ::: "memory");
  __syncthreads();

  const int wave = tid >> 5;
  const int lane = tid & 31;
  const int tile = blockIdx.x * 8 + wave;
  if (tile >= NT) return;
  const int m  = lane & 15;   // node sub-row / D-col index
  const int hi = lane >> 4;   // lane half
  const int nodeBase = tile * 16;

  // Bias values broadcast into C fragments (D layout: col = lane%16)
  float brz[4], bzz[4], bin[4], bhn[4];
#pragma unroll
  for (int ht = 0; ht < 4; ++ht) {
    brz[ht] = b_ih[16 * ht + m] + b_hh[16 * ht + m];
    bzz[ht] = b_ih[64 + 16 * ht + m] + b_hh[64 + 16 * ht + m];
    bin[ht] = b_ih[128 + 16 * ht + m];
    bhn[ht] = b_hh[128 + 16 * ht + m];
  }

  v8f h[4];
#pragma unroll
  for (int ht = 0; ht < 4; ++ht)
#pragma unroll
    for (int e = 0; e < 8; ++e) h[ht][e] = 0.0f;

  _Float16* hrow = &hstage[wave][0];

#pragma unroll 1
  for (int t = 0; t < TS; ++t) {
    // Stage h as f16 row-major [16][64] (D frag: row = e + 8*hi, col = 16*ht+m)
#pragma unroll
    for (int ht = 0; ht < 4; ++ht)
#pragma unroll
      for (int e = 0; e < 8; ++e)
        hrow[(e + hi * 8) * HID + 16 * ht + m] = (_Float16)h[ht][e];
    asm volatile("s_wait_dscnt 0" ::: "memory");  // intra-wave DS RAW fence

    // Build A fragments (16-bit 16x32 layout: lane m holds K = fb..fb+7, fb+16..fb+23)
    ABfrag hA[2], xA[2];
    const _Float16* xrowp = spatial + ((size_t)t * NN + nodeBase + m) * HID;
#pragma unroll
    for (int kc = 0; kc < 2; ++kc) {
      const int fb = kc * 32 + hi * 8;
      hA[kc].h2[0] = *(const v8h*)&hrow[m * HID + fb];
      hA[kc].h2[1] = *(const v8h*)&hrow[m * HID + fb + 16];
      // spatial is read exactly once -> non-temporal (TH=NT) load
      xA[kc].h2[0] = __builtin_nontemporal_load((const v8h*)(xrowp + fb));
      xA[kc].h2[1] = __builtin_nontemporal_load((const v8h*)(xrowp + fb + 16));
    }

#pragma unroll
    for (int ht = 0; ht < 4; ++ht) {
      v8f rp, zp, inp, hnp;
#pragma unroll
      for (int e = 0; e < 8; ++e) {
        rp[e] = brz[ht]; zp[e] = bzz[ht]; inp[e] = bin[ht]; hnp[e] = bhn[ht];
      }

      // Group A: r-gate tiles; Group B: z-gate tiles (disjoint registers)
      ABfrag Ba0 = { load_btile(ldsW, 0, ht, 0, lane) };
      ABfrag Ba1 = { load_btile(ldsW, 0, ht, 1, lane) };
      ABfrag Ba2 = { load_btile(ldsW, 1, ht, 0, lane) };
      ABfrag Ba3 = { load_btile(ldsW, 1, ht, 1, lane) };
      ABfrag Bb0 = { load_btile(ldsW, 0, 4 + ht, 0, lane) };
      ABfrag Bb1 = { load_btile(ldsW, 0, 4 + ht, 1, lane) };
      ABfrag Bb2 = { load_btile(ldsW, 1, 4 + ht, 0, lane) };
      ABfrag Bb3 = { load_btile(ldsW, 1, 4 + ht, 1, lane) };

      // r = sigmoid(x@Wih_r + h@Whh_r + b)
      rp = WMMA(xA[0].v, Ba0.v, rp);
      rp = WMMA(xA[1].v, Ba1.v, rp);
      rp = WMMA(hA[0].v, Ba2.v, rp);
      rp = WMMA(hA[1].v, Ba3.v, rp);

      // reload group A with n-gate tiles while the z chain issues below
      Ba0.v = load_btile(ldsW, 0, 8 + ht, 0, lane);
      Ba1.v = load_btile(ldsW, 0, 8 + ht, 1, lane);
      Ba2.v = load_btile(ldsW, 1, 8 + ht, 0, lane);
      Ba3.v = load_btile(ldsW, 1, 8 + ht, 1, lane);

      // z
      zp = WMMA(xA[0].v, Bb0.v, zp);
      zp = WMMA(xA[1].v, Bb1.v, zp);
      zp = WMMA(hA[0].v, Bb2.v, zp);
      zp = WMMA(hA[1].v, Bb3.v, zp);

      // i_n (x path) and h_n (h path) kept separate: n = tanh(i_n + r*h_n)
      inp = WMMA(xA[0].v, Ba0.v, inp);
      inp = WMMA(xA[1].v, Ba1.v, inp);
      hnp = WMMA(hA[0].v, Ba2.v, hnp);
      hnp = WMMA(hA[1].v, Ba3.v, hnp);

#pragma unroll
      for (int e = 0; e < 8; ++e) {
        float r = fast_sigmoid(rp[e]);
        float z = fast_sigmoid(zp[e]);
        float n = fast_tanh(__builtin_fmaf(r, hnp[e], inp[e]));
        h[ht][e] = (1.0f - z) * n + z * h[ht][e];
      }
    }
  }

  // Epilogue: stage final h, then 64->1 head (lanes 0..15, one node each)
#pragma unroll
  for (int ht = 0; ht < 4; ++ht)
#pragma unroll
    for (int e = 0; e < 8; ++e)
      hrow[(e + hi * 8) * HID + 16 * ht + m] = (_Float16)h[ht][e];
  asm volatile("s_wait_dscnt 0" ::: "memory");
  if (lane < 16) {
    float acc = b_fc[0];
#pragma unroll 8
    for (int j = 0; j < HID; ++j)
      acc += (float)hrow[lane * HID + j] * W_fc[j];
    __builtin_nontemporal_store(acc, &out[nodeBase + lane]);
  }
}

// ---------------------------------------------------------------------------

extern "C" void kernel_launch(void* const* d_in, const int* in_sizes, int n_in,
                              void* d_out, int out_size, void* d_ws, size_t ws_size,
                              hipStream_t stream) {
  (void)in_sizes; (void)n_in; (void)out_size; (void)ws_size;
  const float*     x   = (const float*)d_in[0];
  const long long* ei  = (const long long*)d_in[1];
  const float*     Wg  = (const float*)d_in[2];
  const float*     bg  = (const float*)d_in[3];
  const float*     Wih = (const float*)d_in[4];
  const float*     Whh = (const float*)d_in[5];
  const float*     bih = (const float*)d_in[6];
  const float*     bhh = (const float*)d_in[7];
  const float*     Wfc = (const float*)d_in[8];
  const float*     bfc = (const float*)d_in[9];
  float* out = (float*)d_out;

  char* ws = (char*)d_ws;
  size_t off = 0;
  auto walloc = [&](size_t bytes) -> void* {
    void* p = ws + off;
    off = (off + bytes + 255) & ~(size_t)255;
    return p;
  };
  float*     deg     = (float*)walloc((size_t)NN * 4);            // -> dinv
  float*     agg     = (float*)walloc((size_t)NN * 24 * 4);       // 4.8 MB
  uint32_t*  packedW = (uint32_t*)walloc(12288 * 4);              // 48 KB
  _Float16*  spatial = (_Float16*)walloc((size_t)TS * NN * HID * 2); // 76.8 MB

  k_init   <<<(NN * 24 + 255) / 256, 256, 0, stream>>>(deg, agg);
  k_degree <<<(NE + 255) / 256, 256, 0, stream>>>(ei, deg);
  k_dinv   <<<(NN + 255) / 256, 256, 0, stream>>>(deg);
  k_scatter<<<(NE * 24 + 255) / 256, 256, 0, stream>>>(ei, deg, x, agg);
  k_pack   <<<(12288 + 255) / 256, 256, 0, stream>>>(Wih, Whh, packedW);
  k_spatial<<<(NN * TS * HID + 255) / 256, 256, 0, stream>>>(x, agg, deg, Wg, bg, spatial);
  k_gru    <<<(NT + 7) / 8, 256, 0, stream>>>(spatial, packedW, bih, bhh, Wfc, bfc, out);
}
